// CFGSubASTExpressionCombiner_27857157882342
// MI455X (gfx1250) — compile-verified, matrix operand order checked by reading, not run on hardware
//
#include <hip/hip_runtime.h>
#include <hip/hip_bf16.h>

typedef unsigned short u16;
typedef __attribute__((ext_vector_type(16))) __bf16 bf16x16;
typedef __attribute__((ext_vector_type(8)))  float  f32x8;
typedef __attribute__((ext_vector_type(4)))  float  f32x4;
typedef __attribute__((ext_vector_type(4)))  unsigned int u32x4;

union FragB {
    bf16x16 v;
    u16     s[16];
    u32x4   q[2];
};

__device__ __forceinline__ u16 f2bf(float f) {
    union { float f; unsigned u; } x; x.f = f;
    unsigned r = ((x.u >> 16) & 1u) + 0x7FFFu;   // round-to-nearest-even
    return (u16)((x.u + r) >> 16);
}

__device__ __forceinline__ f32x8 wmma_bf16(bf16x16 a, bf16x16 b, f32x8 c) {
    // D = A(16x32 bf16) * B(32x16 bf16) + C(16x16 f32)
    return __builtin_amdgcn_wmma_f32_16x16x32_bf16(
        false, a, false, b, (short)0, c, false, false);
}

// A fragment (16x32 bf16) from an fp32 row: lane holds row (lane&15),
// halves: i<8 -> K = kt*32 + hb*8 + i ; i>=8 -> K = kt*32 + hb*8 + 16 + (i-8)
__device__ __forceinline__ bf16x16 load_a_f32row(const float* __restrict__ row, int kt, int lane) {
    int hb = (lane >> 4) & 1;
    int c0 = kt * 32 + hb * 8;
    f32x4 f0 = *(const f32x4*)(row + c0);
    f32x4 f1 = *(const f32x4*)(row + c0 + 4);
    f32x4 f2 = *(const f32x4*)(row + c0 + 16);
    f32x4 f3 = *(const f32x4*)(row + c0 + 20);
    FragB fr;
    fr.s[0]  = f2bf(f0[0]); fr.s[1]  = f2bf(f0[1]); fr.s[2]  = f2bf(f0[2]); fr.s[3]  = f2bf(f0[3]);
    fr.s[4]  = f2bf(f1[0]); fr.s[5]  = f2bf(f1[1]); fr.s[6]  = f2bf(f1[2]); fr.s[7]  = f2bf(f1[3]);
    fr.s[8]  = f2bf(f2[0]); fr.s[9]  = f2bf(f2[1]); fr.s[10] = f2bf(f2[2]); fr.s[11] = f2bf(f2[3]);
    fr.s[12] = f2bf(f3[0]); fr.s[13] = f2bf(f3[1]); fr.s[14] = f2bf(f3[2]); fr.s[15] = f2bf(f3[3]);
    return fr.v;
}

// A fragment from a bf16 row (same K mapping), two b128 loads
__device__ __forceinline__ bf16x16 load_a_bf16row(const u16* __restrict__ row, int kt, int lane) {
    int hb = (lane >> 4) & 1;
    int c0 = kt * 32 + hb * 8;
    FragB fr;
    fr.q[0] = *(const u32x4*)(row + c0);
    fr.q[1] = *(const u32x4*)(row + c0 + 16);
    return fr.v;
}

// B fragment from prepacked layout [kt][nt][lane][16] (works for global or LDS ptr)
__device__ __forceinline__ bf16x16 load_b_pack(const u16* __restrict__ base, int kt, int nt, int lane) {
    const u32x4* p = (const u32x4*)(base + (((kt * 16 + nt) * 32 + lane) << 4));
    FragB fr;
    fr.q[0] = p[0];
    fr.q[1] = p[1];
    return fr.v;
}

// ---------------- utility kernels ----------------

__global__ __launch_bounds__(256) void zero_kernel(u32x4* __restrict__ p, long n16) {
    long stride = (long)gridDim.x * blockDim.x;
    for (long i = (long)blockIdx.x * blockDim.x + threadIdx.x; i < n16; i += stride) {
        u32x4 z = {0u, 0u, 0u, 0u};
        p[i] = z;
    }
}

__global__ __launch_bounds__(256) void init_owner_kernel(int* __restrict__ owner, int C) {
    int i = blockIdx.x * blockDim.x + threadIdx.x;
    if (i < C) owner[i] = -1;
}

// Pack 4 weight matrices (256x256 fp32, row=K, col=N) into WMMA B-fragment bf16 layout
__global__ __launch_bounds__(256) void prepack_kernel(const float* __restrict__ Wq,
                                                      const float* __restrict__ Wk,
                                                      const float* __restrict__ Wv,
                                                      const float* __restrict__ Wo,
                                                      u16* __restrict__ packs) {
    int t = blockIdx.x * blockDim.x + threadIdx.x;
    if (t >= 4 * 8 * 16 * 32) return;
    int lane = t & 31;
    int nt   = (t >> 5) & 15;
    int kt   = (t >> 9) & 7;
    int w    = t >> 12;
    const float* W = (w == 0) ? Wq : (w == 1) ? Wk : (w == 2) ? Wv : Wo;
    u16* dst = packs + (long)w * 65536 + (((kt * 16 + nt) * 32 + lane) << 4);
    int hb = lane >> 4, n = lane & 15;
    #pragma unroll
    for (int i = 0; i < 16; ++i) {
        int kl = hb * 8 + (i < 8 ? i : i + 8);
        dst[i] = f2bf(W[(kt * 32 + kl) * 256 + nt * 16 + n]);
    }
}

// deterministic scatter: highest c wins per key
__global__ __launch_bounds__(256) void scatter_owner_kernel(const int* __restrict__ key,
                                                            int* __restrict__ owner, int C) {
    int c = blockIdx.x * blockDim.x + threadIdx.x;
    if (c < C) atomicMax(&owner[key[c]], c);
}

__global__ __launch_bounds__(256) void scatter_write_kernel(const float* __restrict__ ast,
                                                            const int* __restrict__ key,
                                                            const int* __restrict__ val,
                                                            const int* __restrict__ owner,
                                                            u16* __restrict__ attnbf, int C) {
    long t = (long)blockIdx.x * blockDim.x + threadIdx.x;
    long total = (long)C * 256;
    if (t >= total) return;
    int c = (int)(t >> 8);
    int col = (int)(t & 255);
    int k = key[c];
    if (owner[k] != c) return;
    float x = ast[(long)val[c] * 256 + col];
    attnbf[(long)k * 256 + col] = f2bf(x);
}

// ---------------- q = (attn_keys @ Wq + bq) * (1/sqrt(64)) ----------------

__global__ __launch_bounds__(256) void q_gemm_kernel(const u16* __restrict__ attnbf,
                                                     const u16* __restrict__ wqpack,
                                                     const float* __restrict__ bq,
                                                     float* __restrict__ qout, int C) {
    int lane = threadIdx.x & 31;
    int wave = threadIdx.x >> 5;
    int ntiles = (C + 15) >> 4;
    int tile = blockIdx.x * 8 + wave;
    if (tile >= ntiles) return;
    int rbase = tile * 16;
    int n = lane & 15, hb = lane >> 4;
    bool full = (rbase + 16 <= C);

    int ar = rbase + n; if (ar >= C) ar = C - 1;
    const u16* row = attnbf + (long)ar * 256;
    bf16x16 afr[8];
    #pragma unroll
    for (int kt = 0; kt < 8; ++kt) afr[kt] = load_a_bf16row(row, kt, lane);

    for (int nt = 0; nt < 16; ++nt) {
        f32x8 c = {};
        #pragma unroll
        for (int kt = 0; kt < 8; ++kt)
            c = wmma_bf16(afr[kt], load_b_pack(wqpack, kt, nt, lane), c);
        float bqv = bq[nt * 16 + n];
        float* o = qout + (long)(rbase + hb * 8) * 256 + nt * 16 + n;
        if (full) {
            #pragma unroll
            for (int j = 0; j < 8; ++j) o[(long)j * 256] = (c[j] + bqv) * 0.125f;
        } else {
            #pragma unroll
            for (int j = 0; j < 8; ++j) {
                int r = rbase + j + hb * 8;
                if (r < C) qout[(long)r * 256 + nt * 16 + n] = (c[j] + bqv) * 0.125f;
            }
        }
    }
}

// ---------------- fused K/V + segment softmax-pool ----------------
// per wave: 16 rows of the scattered matrix; two phases, each staging one
// 128 KB weight-fragment pack in LDS (2 workgroups/WGP for latency hiding)

__global__ __launch_bounds__(256) void attn_main_kernel(const float* __restrict__ ast,
                                                        const int* __restrict__ astkey,
                                                        const int* __restrict__ segv,
                                                        const float* __restrict__ qbuf,
                                                        const u16* __restrict__ wkpack,
                                                        const u16* __restrict__ wvpack,
                                                        const float* __restrict__ bk,
                                                        const float* __restrict__ bv,
                                                        float* __restrict__ num,
                                                        float* __restrict__ denom,
                                                        int M) {
    extern __shared__ u16 lds[];   // 65536 u16 = 128 KB (one weight pack)

    int lane = threadIdx.x & 31;
    int wave = threadIdx.x >> 5;
    int ntiles = (M + 15) >> 4;
    int tile = blockIdx.x * 8 + wave;
    bool active = (tile < ntiles);
    int mbase = active ? tile * 16 : 0;
    int n = lane & 15, hb = lane >> 4;

    // gather this lane's A row (harmless dummy work for inactive tail waves)
    int am = mbase + n; if (am >= M) am = M - 1;
    const float* xrow = ast + (long)astkey[am] * 256;
    bf16x16 afr[8];
    #pragma unroll
    for (int kt = 0; kt < 8; ++kt) afr[kt] = load_a_f32row(xrow, kt, lane);

    // segment ids + validity for the 8 output rows this lane touches
    int  seg_j[8];
    bool ok_j[8];
    #pragma unroll
    for (int j = 0; j < 8; ++j) {
        int r = mbase + j + hb * 8;
        ok_j[j] = active && (r < M);
        seg_j[j] = segv[(r < M) ? r : (M - 1)];
    }

    // ---- phase 1: stage Wk pack, compute scores + exp + denom ----
    {
        u32x4* d = (u32x4*)lds;
        const u32x4* s = (const u32x4*)wkpack;
        for (int i = threadIdx.x; i < 8192; i += 256) d[i] = s[i];
    }
    __syncthreads();

    float e[4][8];
    if (active) {
        for (int h = 0; h < 4; ++h) {
            float acc[8];
            #pragma unroll
            for (int j = 0; j < 8; ++j) acc[j] = 0.f;

            #pragma unroll
            for (int nt4 = 0; nt4 < 4; ++nt4) {
                int nt = h * 4 + nt4;
                f32x8 c = {};
                #pragma unroll
                for (int kt = 0; kt < 8; ++kt)
                    c = wmma_bf16(afr[kt], load_b_pack(lds, kt, nt, lane), c);
                float bkv = bk[nt * 16 + n];
                #pragma unroll
                for (int j = 0; j < 8; ++j) {
                    float qv = qbuf[(long)seg_j[j] * 256 + nt * 16 + n];
                    acc[j] += (c[j] + bkv) * qv;   // scale already folded into q
                }
            }
            // reduce across the 16 lanes of each half-wave; every lane gets the sum
            #pragma unroll
            for (int j = 0; j < 8; ++j) {
                float v = acc[j];
                v += __shfl_xor(v, 1, 16);
                v += __shfl_xor(v, 2, 16);
                v += __shfl_xor(v, 4, 16);
                v += __shfl_xor(v, 8, 16);
                e[h][j] = __expf(v);   // logits are O(0.1); max-shift omitted (exact in exact arith)
            }
            if (n == 0) {
                #pragma unroll
                for (int j = 0; j < 8; ++j)
                    if (ok_j[j]) atomicAdd(&denom[(long)seg_j[j] * 4 + h], e[h][j]);
            }
        }
    }

    // ---- phase 2: restage with Wv pack, weighted accumulate into num[seg] ----
    __syncthreads();
    {
        u32x4* d = (u32x4*)lds;
        const u32x4* s = (const u32x4*)wvpack;
        for (int i = threadIdx.x; i < 8192; i += 256) d[i] = s[i];
    }
    __syncthreads();

    if (active) {
        for (int h = 0; h < 4; ++h) {
            #pragma unroll
            for (int nt4 = 0; nt4 < 4; ++nt4) {
                int nt = h * 4 + nt4;
                f32x8 c = {};
                #pragma unroll
                for (int kt = 0; kt < 8; ++kt)
                    c = wmma_bf16(afr[kt], load_b_pack(lds, kt, nt, lane), c);
                float bvv = bv[nt * 16 + n];
                #pragma unroll
                for (int j = 0; j < 8; ++j) {
                    if (ok_j[j])
                        atomicAdd(&num[(long)seg_j[j] * 256 + nt * 16 + n],
                                  e[h][j] * (c[j] + bvv));
                }
            }
        }
    }
}

// ---------------- out = (num/denom) @ Wo + bo ----------------

__global__ __launch_bounds__(256) void out_gemm_kernel(const float* __restrict__ num,
                                                       const float* __restrict__ denom,
                                                       const u16* __restrict__ wopack,
                                                       const float* __restrict__ bo,
                                                       float* __restrict__ out, int C) {
    int lane = threadIdx.x & 31;
    int wave = threadIdx.x >> 5;
    int ntiles = (C + 15) >> 4;
    int tile = blockIdx.x * 8 + wave;
    if (tile >= ntiles) return;
    int rbase = tile * 16;
    int n = lane & 15, hb = lane >> 4;
    bool full = (rbase + 16 <= C);

    int ar = rbase + n; if (ar >= C) ar = C - 1;
    const float* nrow = num + (long)ar * 256;
    const float* drow = denom + (long)ar * 4;
    float dinv[4];
    #pragma unroll
    for (int hh = 0; hh < 4; ++hh) dinv[hh] = 1.0f / (drow[hh] + 1e-9f);

    bf16x16 afr[8];
    #pragma unroll
    for (int kt = 0; kt < 8; ++kt) {
        int c0 = kt * 32 + hb * 8;
        FragB fr;
        #pragma unroll
        for (int i = 0; i < 8; ++i) {
            int col = c0 + i;
            fr.s[i] = f2bf(nrow[col] * dinv[col >> 6]);
        }
        #pragma unroll
        for (int i = 8; i < 16; ++i) {
            int col = c0 + 16 + (i - 8);
            fr.s[i] = f2bf(nrow[col] * dinv[col >> 6]);
        }
        afr[kt] = fr.v;
    }

    for (int nt = 0; nt < 16; ++nt) {
        f32x8 c = {};
        #pragma unroll
        for (int kt = 0; kt < 8; ++kt)
            c = wmma_bf16(afr[kt], load_b_pack(wopack, kt, nt, lane), c);
        float bov = bo[nt * 16 + n];
        float* o = out + (long)(rbase + hb * 8) * 256 + nt * 16 + n;
        if (full) {
            #pragma unroll
            for (int j = 0; j < 8; ++j) o[(long)j * 256] = c[j] + bov;
        } else {
            #pragma unroll
            for (int j = 0; j < 8; ++j) {
                int r = rbase + j + hb * 8;
                if (r < C) out[(long)r * 256 + nt * 16 + n] = c[j] + bov;
            }
        }
    }
}

// ---------------- host launch ----------------

extern "C" void kernel_launch(void* const* d_in, const int* in_sizes, int n_in,
                              void* d_out, int out_size, void* d_ws, size_t ws_size,
                              hipStream_t stream) {
    (void)n_in; (void)out_size; (void)ws_size;
    const float* ast    = (const float*)d_in[0];
    const int*   astkey = (const int*)d_in[1];   // [M] -> ast row
    const int*   segv   = (const int*)d_in[2];   // [M] -> segment (pdg node)
    const int*   pdgkey = (const int*)d_in[3];   // [C]
    const int*   pdgval = (const int*)d_in[4];   // [C] -> ast row
    // d_in[5] = nr_cfg_nodes scalar (use in_sizes instead)
    const float* Wq = (const float*)d_in[6];
    const float* bq = (const float*)d_in[7];
    const float* Wk = (const float*)d_in[8];
    const float* bk = (const float*)d_in[9];
    const float* Wv = (const float*)d_in[10];
    const float* bv = (const float*)d_in[11];
    const float* Wo = (const float*)d_in[12];
    const float* bo = (const float*)d_in[13];
    float* out = (float*)d_out;

    const int M = in_sizes[1];
    const int C = in_sizes[3];

    auto al = [](size_t x) { return (x + 255) & ~(size_t)255; };
    char* w = (char*)d_ws;
    size_t o_attn = 0;
    size_t o_num  = o_attn + al((size_t)C * 256 * 2);
    size_t o_den  = o_num  + al((size_t)C * 256 * 4);
    size_t o_zend = o_den  + al((size_t)C * 4 * 4);
    size_t o_q    = o_zend;
    size_t o_pack = o_q    + al((size_t)C * 256 * 4);
    size_t o_own  = o_pack + al((size_t)4 * 65536 * 2);

    u16*   attnbf = (u16*)(w + o_attn);
    float* numb   = (float*)(w + o_num);
    float* denb   = (float*)(w + o_den);
    float* qbuf   = (float*)(w + o_q);
    u16*   packs  = (u16*)(w + o_pack);
    int*   owner  = (int*)(w + o_own);

    u16* wqpack = packs;
    u16* wkpack = packs + 65536;
    u16* wvpack = packs + 2 * 65536;
    u16* wopack = packs + 3 * 65536;

    // 1) zero accumulators (attn_keys_bf + num + denom, contiguous span)
    long n16 = (long)(o_zend / 16);
    zero_kernel<<<dim3(2048), dim3(256), 0, stream>>>((u32x4*)w, n16);

    // 2) owner init + weight prepack
    init_owner_kernel<<<dim3((C + 255) / 256), dim3(256), 0, stream>>>(owner, C);
    prepack_kernel<<<dim3(64), dim3(256), 0, stream>>>(Wq, Wk, Wv, Wo, packs);

    // 3) deterministic scatter of root encodings into attn_keys (bf16)
    scatter_owner_kernel<<<dim3((C + 255) / 256), dim3(256), 0, stream>>>(pdgkey, owner, C);
    long scat_threads = (long)C * 256;
    scatter_write_kernel<<<dim3((unsigned)((scat_threads + 255) / 256)), dim3(256), 0, stream>>>(
        ast, pdgkey, pdgval, owner, attnbf, C);

    // 4) q projection (bias + 1/sqrt(d_h) folded)
    int ntC = (C + 15) >> 4;
    q_gemm_kernel<<<dim3((ntC + 7) / 8), dim3(256), 0, stream>>>(attnbf, wqpack, bq, qbuf, C);

    // 5) fused K/V + segment softmax pooling (128 KB dynamic LDS, two phases)
    int ntM = (M + 15) >> 4;
    attn_main_kernel<<<dim3((ntM + 7) / 8), dim3(256), 131072, stream>>>(
        ast, astkey, segv, qbuf, wkpack, wvpack, bk, bv, numb, denb, M);

    // 6) output projection
    out_gemm_kernel<<<dim3((ntC + 7) / 8), dim3(256), 0, stream>>>(numb, denb, wopack, bo, out, C);
}
